// mamba2_time_series_19550691131719
// MI455X (gfx1250) — compile-verified, matrix operand order checked by reading, not run on hardware
//
#include <hip/hip_runtime.h>
#include <hip/hip_bf16.h>
#include <math.h>

typedef __attribute__((ext_vector_type(16))) _Float16 v16h;
typedef __attribute__((ext_vector_type(8)))  _Float16 v8h;
typedef __attribute__((ext_vector_type(8)))  float    v8f;

#define D_MODEL   768
#define N_HEADS   24
#define HEAD_DIM  64
#define D_INNER   1536
#define D_STATE   128
#define CONV_K    4
#define CONV_DIM  1792            // D_INNER + 2*D_STATE
#define PROJ_DIM  3352            // 2*D_INNER + 2*D_STATE + N_HEADS
#define CHUNKSZ   256
#define N_LAYERS  24
#define BATCH     2
#define SEQ       1024
#define ROWS      (BATCH*SEQ)     // 2048
#define NCHUNK    (SEQ/CHUNKSZ)   // 4
#define EPSV      1e-5f

// ---------------------------------------------------------------------------
// device helpers
// ---------------------------------------------------------------------------
__device__ __forceinline__ float gelu_f(float x) {
    return 0.5f * x * (1.0f + erff(x * 0.70710678118654752f));
}
__device__ __forceinline__ float silu_f(float x) {
    return x / (1.0f + expf(-x));
}

// ---------------------------------------------------------------------------
// Batched WMMA GEMM:  D[M,N] = A[M,K] * B[N,K]^T   (f16 in, f32 accumulate)
//   - every operand is K-contiguous => all staging is 16B global_load_b128.
//   - 256 threads (8 waves); block tile 128(M) x 64(N); K stepped by 64.
//   - each wave owns four 16x16 accumulators => 8 v_wmma per K-iteration.
//   - A tile (128x64) and B tile (64x64) staged in LDS, both K-major, so
//     per-lane WMMA gathers are contiguous ds_load_b128 pairs.
//   - maskMode: D'[k][s] = (k>=s) ? exp(acs[k]-acs[s]) * D : 0, written f16.
//   - rowScale: D *= rowScale[row] before optional accumulate into f32 D.
// ---------------------------------------------------------------------------
struct GemmP {
    const _Float16* A; const _Float16* B;
    float* D; _Float16* Dh;
    const float* rowScale; const float* acs;
    int M, N, K, lda, ldb, ldd;
    int accum, maskMode;
    int nC, nH;
    long long sAb, sAc, sAh;
    long long sBb, sBc, sBh;
    long long sDb, sDc, sDh;
    long long sRb, sRc, sRh;
};

__global__ __launch_bounds__(256) void k_gemm(GemmP p) {
    __shared__ _Float16 lA[128 * 64];   // A tile, M-major rows (stride 64 halves = 128B)
    __shared__ _Float16 lB[64 * 64];    // B tile, N-major rows (stride 64 halves)

    const int tid  = threadIdx.x;
    const int lane = tid & 31;
    const int wave = tid >> 5;
    const int n0   = blockIdx.x * 64;
    const int m0   = blockIdx.y * 128;

    const int z  = blockIdx.z;
    const int hh = z % p.nH;
    const int cc = (z / p.nH) % p.nC;
    const int bb = z / (p.nH * p.nC);

    const long long aBase = (long long)bb * p.sAb + (long long)cc * p.sAc + (long long)hh * p.sAh;
    const long long bBase = (long long)bb * p.sBb + (long long)cc * p.sBc + (long long)hh * p.sBh;
    const long long dBase = (long long)bb * p.sDb + (long long)cc * p.sDc + (long long)hh * p.sDh;
    const long long rBase = (long long)bb * p.sRb + (long long)cc * p.sRc + (long long)hh * p.sRh;

    const int hi = lane >> 4;        // lanes 16..31 hold the "high" K halves
    const int mr = lane & 15;

    v8f acc0 = {}, acc1 = {}, acc2 = {}, acc3 = {};

    for (int k0 = 0; k0 < p.K; k0 += 64) {
        __syncthreads();
        // ---- stage A tile (128 rows x 64 k): 1024 v8h, 4 per thread ----
#pragma unroll
        for (int i = 0; i < 4; ++i) {
            const int v  = tid + 256 * i;
            const int r  = v >> 3, seg = v & 7;
            const int gm = m0 + r;
            v8h val = {};
            if (gm < p.M) {
                const _Float16* ap = p.A + aBase + (long long)gm * p.lda + (k0 + seg * 8);
                val = *(const v8h*)ap;
                if (k0 + 64 < p.K) __builtin_prefetch(ap + 64, 0, 0);
            }
            ((v8h*)lA)[v] = val;
        }
        // ---- stage B tile (64 rows x 64 k): 512 v8h, 2 per thread ----
#pragma unroll
        for (int i = 0; i < 2; ++i) {
            const int v  = tid + 256 * i;
            const int n  = v >> 3, seg = v & 7;
            const int gn = n0 + n;
            v8h val = {};
            if (gn < p.N)
                val = *(const v8h*)(p.B + bBase + (long long)gn * p.ldb + (k0 + seg * 8));
            ((v8h*)lB)[v] = val;
        }
        __syncthreads();

        // ---- two k-substeps of 32, four N-tiles each: 8 WMMA ----
#pragma unroll
        for (int ks = 0; ks < 2; ++ks) {
            const v8h* ap = (const v8h*)(lA + (wave * 16 + mr) * 64 + ks * 32);
            v8h a0 = ap[hi], a1 = ap[2 + hi];
            v16h av = __builtin_shufflevector(a0, a1, 0,1,2,3,4,5,6,7,8,9,10,11,12,13,14,15);

            const v8h* bp0 = (const v8h*)(lB + (0 * 16 + mr) * 64 + ks * 32);
            const v8h* bp1 = (const v8h*)(lB + (1 * 16 + mr) * 64 + ks * 32);
            const v8h* bp2 = (const v8h*)(lB + (2 * 16 + mr) * 64 + ks * 32);
            const v8h* bp3 = (const v8h*)(lB + (3 * 16 + mr) * 64 + ks * 32);
            v8h b0l = bp0[hi], b0h = bp0[2 + hi];
            v8h b1l = bp1[hi], b1h = bp1[2 + hi];
            v8h b2l = bp2[hi], b2h = bp2[2 + hi];
            v8h b3l = bp3[hi], b3h = bp3[2 + hi];
            v16h bv0 = __builtin_shufflevector(b0l, b0h, 0,1,2,3,4,5,6,7,8,9,10,11,12,13,14,15);
            v16h bv1 = __builtin_shufflevector(b1l, b1h, 0,1,2,3,4,5,6,7,8,9,10,11,12,13,14,15);
            v16h bv2 = __builtin_shufflevector(b2l, b2h, 0,1,2,3,4,5,6,7,8,9,10,11,12,13,14,15);
            v16h bv3 = __builtin_shufflevector(b3l, b3h, 0,1,2,3,4,5,6,7,8,9,10,11,12,13,14,15);

            acc0 = __builtin_amdgcn_wmma_f32_16x16x32_f16(false, av, false, bv0, (short)0, acc0, false, false);
            acc1 = __builtin_amdgcn_wmma_f32_16x16x32_f16(false, av, false, bv1, (short)0, acc1, false, false);
            acc2 = __builtin_amdgcn_wmma_f32_16x16x32_f16(false, av, false, bv2, (short)0, acc2, false, false);
            acc3 = __builtin_amdgcn_wmma_f32_16x16x32_f16(false, av, false, bv3, (short)0, acc3, false, false);
        }
    }

    // ---- epilogue: D[m = wave*16 + v + 8*hi][n] for four N-tiles ----
#pragma unroll
    for (int tile = 0; tile < 4; ++tile) {
        const int col = n0 + tile * 16 + mr;
        if (col >= p.N) continue;
        const v8f& acc = (tile == 0) ? acc0 : (tile == 1) ? acc1 : (tile == 2) ? acc2 : acc3;
#pragma unroll
        for (int v = 0; v < 8; ++v) {
            const int gm = m0 + wave * 16 + v + 8 * hi;
            if (gm >= p.M) continue;
            float val = acc[v];
            if (p.maskMode) {
                val = (gm >= col) ? val * expf(p.acs[rBase + gm] - p.acs[rBase + col]) : 0.f;
            } else if (p.rowScale) {
                val *= p.rowScale[rBase + gm];
            }
            const long long di = dBase + (long long)gm * p.ldd + col;
            if (p.Dh)            p.Dh[di] = (_Float16)val;
            else if (p.accum)    p.D[di] += val;
            else                 p.D[di]  = val;
        }
    }
}

// ---------------------------------------------------------------------------
// f32 [K,N] -> f16 [N,K] transposed conversion (weights)
// ---------------------------------------------------------------------------
__global__ void k_cvtT(const float* __restrict__ src, _Float16* __restrict__ dst,
                       int K, int N) {
    long i = (long)blockIdx.x * blockDim.x + threadIdx.x;
    if (i >= (long)K * N) return;
    int k = (int)(i % K), n = (int)(i / K);
    dst[i] = (_Float16)src[(long)k * N + n];
}

// ---------------------------------------------------------------------------
// stem convolutions (gelu + batchnorm folded)
// ---------------------------------------------------------------------------
__global__ void k_conv1(const float* x, const float* w, const float* bias,
                        const float* g, const float* bb, const float* m, const float* var,
                        float* out) {
    int idx = blockIdx.x * blockDim.x + threadIdx.x;
    if (idx >= 2 * 192 * 2048) return;
    int t = idx % 2048, oc = (idx / 2048) % 192, b = idx / (2048 * 192);
    float acc = bias[oc];
    for (int c = 0; c < 12; ++c)
        for (int j = 0; j < 5; ++j) {
            int ip = 2 * t + j - 2;
            if (ip >= 0 && ip < 4096)
                acc += w[(oc * 12 + c) * 5 + j] * x[((size_t)b * 12 + c) * 4096 + ip];
        }
    acc = gelu_f(acc);
    out[idx] = (acc - m[oc]) * (g[oc] * rsqrtf(var[oc] + EPSV)) + bb[oc];
}

__global__ void k_conv2(const float* in, const float* w, const float* bias,
                        const float* g, const float* bb, const float* m, const float* var,
                        float* out) {
    int idx = blockIdx.x * blockDim.x + threadIdx.x;
    if (idx >= 2 * 384 * 1024) return;
    int t = idx % 1024, oc = (idx / 1024) % 384, b = idx / (1024 * 384);
    float acc = bias[oc];
    for (int c = 0; c < 192; ++c)
        for (int j = 0; j < 3; ++j) {
            int ip = 2 * t + j - 1;
            if (ip >= 0 && ip < 2048)
                acc += w[(oc * 192 + c) * 3 + j] * in[((size_t)b * 192 + c) * 2048 + ip];
        }
    acc = gelu_f(acc);
    out[idx] = (acc - m[oc]) * (g[oc] * rsqrtf(var[oc] + EPSV)) + bb[oc];
}

__global__ void k_conv3(const float* in, const float* w, const float* bias,
                        const float* g, const float* bb, const float* m, const float* var,
                        float* hout) {   // writes transposed: h[(b*1024+t)*768 + oc]
    int idx = blockIdx.x * blockDim.x + threadIdx.x;
    if (idx >= 2 * 768 * 1024) return;
    int t = idx % 1024, oc = (idx / 1024) % 768, b = idx / (1024 * 768);
    float acc = bias[oc];
    for (int c = 0; c < 384; ++c)
        for (int j = 0; j < 3; ++j) {
            int ip = t + j - 1;
            if (ip >= 0 && ip < 1024)
                acc += w[(oc * 384 + c) * 3 + j] * in[((size_t)b * 384 + c) * 1024 + ip];
        }
    acc = gelu_f(acc);
    acc = (acc - m[oc]) * (g[oc] * rsqrtf(var[oc] + EPSV)) + bb[oc];
    hout[((size_t)b * 1024 + t) * 768 + oc] = acc;
}

// ---------------------------------------------------------------------------
// RMS norm row kernel (f32 in -> f16 out), block per row
// ---------------------------------------------------------------------------
__global__ __launch_bounds__(256) void k_rmsrow(const float* in, const float* w,
                                                _Float16* out, int width) {
    __shared__ float red[256];
    int row = blockIdx.x, t = threadIdx.x;
    const float* rp = in + (size_t)row * width;
    float ss = 0.f;
    for (int d = t; d < width; d += 256) { float v = rp[d]; ss += v * v; }
    red[t] = ss; __syncthreads();
    for (int o = 128; o > 0; o >>= 1) { if (t < o) red[t] += red[t + o]; __syncthreads(); }
    float sc = rsqrtf(red[0] / (float)width + EPSV);
    for (int d = t; d < width; d += 256)
        out[(size_t)row * width + d] = (_Float16)(rp[d] * sc * w[d]);
}

// ---------------------------------------------------------------------------
// dt = softplus(proj_dt + dt_bias); dA = dt * (-exp(A_log))
// ---------------------------------------------------------------------------
__global__ void k_postproj(const float* proj, const float* dtb, const float* alog,
                           float* dtbuf, float* dAbuf) {
    int idx = blockIdx.x * blockDim.x + threadIdx.x;
    if (idx >= ROWS * N_HEADS) return;
    int row = idx / N_HEADS, hh = idx % N_HEADS;
    float v = proj[(size_t)row * PROJ_DIM + (D_INNER + CONV_DIM) + hh] + dtb[hh];
    float dt = (v > 20.f) ? v : log1pf(expf(v));
    dtbuf[idx] = dt;
    int b = row / SEQ, l = row % SEQ;
    dAbuf[((size_t)(b * N_HEADS + hh)) * SEQ + l] = dt * (-expf(alog[hh]));
}

// ---------------------------------------------------------------------------
// per-(b,h,chunk) inclusive cumsum of dA (Hillis-Steele over 256)
// -> Acs, exp(Acs), decay_states = exp(last - Acs), chunk_decay = exp(last)
// ---------------------------------------------------------------------------
__global__ __launch_bounds__(256) void k_cumsum(const float* dA, float* Acs, float* expA,
                                                float* decay, float* cdecay) {
    __shared__ float s[256];
    int z = blockIdx.x;              // 0 .. 2*24*4-1
    int c = z % NCHUNK, bh = z / NCHUNK;
    int t = threadIdx.x;
    size_t base = (size_t)bh * SEQ + (size_t)c * CHUNKSZ;
    s[t] = dA[base + t];
    __syncthreads();
    for (int off = 1; off < 256; off <<= 1) {
        float add = (t >= off) ? s[t - off] : 0.f;
        __syncthreads();
        s[t] += add;
        __syncthreads();
    }
    float a = s[t], last = s[255];
    Acs[base + t]   = a;
    expA[base + t]  = expf(a);
    decay[base + t] = expf(last - a);
    if (t == 0) cdecay[bh * NCHUNK + c] = expf(last);
}

// ---------------------------------------------------------------------------
// depthwise causal conv (K=4) + silu over xBC columns of proj
// ---------------------------------------------------------------------------
__global__ void k_dwconv(const float* proj, const float* cw, const float* cb, float* xBCc) {
    int idx = blockIdx.x * blockDim.x + threadIdx.x;
    if (idx >= ROWS * CONV_DIM) return;
    int ch = idx % CONV_DIM, row = idx / CONV_DIM;
    int b = row / SEQ, l = row % SEQ;
    float acc = cb[ch];
    for (int j = 0; j < CONV_K; ++j) {
        int lp = l - (CONV_K - 1) + j;
        if (lp >= 0)
            acc += cw[ch * CONV_K + j] * proj[((size_t)(b * SEQ + lp)) * PROJ_DIM + D_INNER + ch];
    }
    xBCc[idx] = silu_f(acc);
}

// ---------------------------------------------------------------------------
// pack f16 operands, K-contiguous layouts for all GEMM uses:
//   XdtT[(b*24+h)*64+p][l] = X*dt          (Ydiag B operand, N=p rows)
//   XwT [(b*24+h)*64+p][l] = X*dt*decay    (states A operand, M=p rows)
//   Bf16[b*1024+l][n]                      (S  B operand, N=s rows, K=n)
//   BfT [b*128+n][l]                       (states B operand, N=n rows)
//   Cf16[b*1024+l][n]                      (S/Yoff A operand, M=k rows, K=n)
// ---------------------------------------------------------------------------
__global__ void k_pack(const float* xBCc, const float* dtbuf, const float* decay,
                       _Float16* XdtT, _Float16* XwT, _Float16* Bf, _Float16* BfT,
                       _Float16* Cf) {
    int idx = blockIdx.x * blockDim.x + threadIdx.x;
    if (idx >= ROWS * CONV_DIM) return;
    int ch = idx % CONV_DIM, row = idx / CONV_DIM;
    int b = row / SEQ, l = row % SEQ;
    float v = xBCc[idx];
    if (ch < D_INNER) {
        int hh = ch >> 6, pp = ch & 63;
        float dt = dtbuf[row * N_HEADS + hh];
        float xd = v * dt;
        size_t ti = ((size_t)(b * N_HEADS + hh) * HEAD_DIM + pp) * SEQ + l;
        XdtT[ti] = (_Float16)xd;
        XwT[ti]  = (_Float16)(xd * decay[((size_t)(b * N_HEADS + hh)) * SEQ + l]);
    } else if (ch < D_INNER + D_STATE) {
        int n = ch - D_INNER;
        Bf[(size_t)row * D_STATE + n] = (_Float16)v;
        BfT[((size_t)b * D_STATE + n) * SEQ + l] = (_Float16)v;
    } else {
        Cf[(size_t)row * D_STATE + (ch - D_INNER - D_STATE)] = (_Float16)v;
    }
}

// ---------------------------------------------------------------------------
// inter-chunk scan: prev[c] = carry; carry = carry * chunk_decay[c] + states[c]
// grid = 48 blocks (b*24+h), 256 threads; 64*128 = 8192 elements per state
// ---------------------------------------------------------------------------
__global__ __launch_bounds__(256) void k_scan(const float* states, _Float16* prevf,
                                              const float* cdecay) {
    int bh = blockIdx.x;                // b*24 + h
    int b = bh / N_HEADS, hh = bh % N_HEADS;
    const int PER = 8192 / 256;         // 32
    float carry[PER];
#pragma unroll
    for (int i = 0; i < PER; ++i) carry[i] = 0.f;
    for (int c = 0; c < NCHUNK; ++c) {
        size_t base = (((size_t)b * NCHUNK + c) * N_HEADS + hh) * 8192;
        float cd = cdecay[bh * NCHUNK + c];
        for (int i = 0; i < PER; ++i) {
            int e = i * 256 + threadIdx.x;
            prevf[base + e] = (_Float16)carry[i];
            carry[i] = carry[i] * cd + states[base + e];
        }
    }
}

// ---------------------------------------------------------------------------
// gating: y = rms((Y + D_skip*X) * silu(z), gn_w) -> f16
// ---------------------------------------------------------------------------
__global__ __launch_bounds__(256) void k_gate(const float* proj, const float* Y,
                                              const float* xBCc, const float* dsk,
                                              const float* gnw, _Float16* out) {
    __shared__ float g[D_INNER];
    __shared__ float red[256];
    int row = blockIdx.x, t = threadIdx.x;
    float ss = 0.f;
    for (int d = t; d < D_INNER; d += 256) {
        float z  = proj[(size_t)row * PROJ_DIM + d];
        float yv = Y[(size_t)row * D_INNER + d] + dsk[d >> 6] * xBCc[(size_t)row * CONV_DIM + d];
        float gv = yv * silu_f(z);
        g[d] = gv; ss += gv * gv;
    }
    red[t] = ss; __syncthreads();
    for (int o = 128; o > 0; o >>= 1) { if (t < o) red[t] += red[t + o]; __syncthreads(); }
    float sc = rsqrtf(red[0] / (float)D_INNER + EPSV);
    for (int d = t; d < D_INNER; d += 256)
        out[(size_t)row * D_INNER + d] = (_Float16)(g[d] * sc * gnw[d]);
}

// ---------------------------------------------------------------------------
// final: per-row rms scale, pooled mean, FC head
// ---------------------------------------------------------------------------
__global__ __launch_bounds__(256) void k_rowr(const float* in, float* rowr, int width) {
    __shared__ float red[256];
    int row = blockIdx.x, t = threadIdx.x;
    const float* rp = in + (size_t)row * width;
    float ss = 0.f;
    for (int d = t; d < width; d += 256) { float v = rp[d]; ss += v * v; }
    red[t] = ss; __syncthreads();
    for (int o = 128; o > 0; o >>= 1) { if (t < o) red[t] += red[t + o]; __syncthreads(); }
    if (t == 0) rowr[row] = rsqrtf(red[0] / (float)width + EPSV);
}

__global__ void k_pool(const float* h, const float* rowr, const float* normf, float* pooled) {
    int idx = blockIdx.x * blockDim.x + threadIdx.x;
    if (idx >= BATCH * D_MODEL) return;
    int b = idx / D_MODEL, d = idx % D_MODEL;
    float s = 0.f;
    for (int l = 0; l < SEQ; ++l) {
        int r = b * SEQ + l;
        s += h[(size_t)r * D_MODEL + d] * rowr[r];
    }
    pooled[idx] = s * (1.f / (float)SEQ) * normf[d];
}

__global__ __launch_bounds__(512) void k_head(const float* pooled,
                                              const float* w1, const float* b1,
                                              const float* g, const float* bb,
                                              const float* m, const float* var,
                                              const float* w2, const float* b2, float* out) {
    __shared__ float f[512];
    int b = blockIdx.x, t = threadIdx.x;
    float acc = b1[t];
    for (int k = 0; k < D_MODEL; ++k) acc += pooled[b * D_MODEL + k] * w1[k * 512 + t];
    acc = (acc - m[t]) * (g[t] * rsqrtf(var[t] + EPSV)) + bb[t];
    f[t] = acc > 0.f ? acc : 0.f;
    __syncthreads();
    if (t < 27) {
        float o = b2[t];
        for (int k = 0; k < 512; ++k) o += f[k] * w2[k * 27 + t];
        out[b * 27 + t] = o;
    }
}

// ---------------------------------------------------------------------------
// host orchestration
// ---------------------------------------------------------------------------
static inline char* carve(char*& p, size_t bytes) {
    char* r = p;
    p += (bytes + 255) & ~(size_t)255;
    return r;
}

static inline void launch_gemm(const GemmP& p, hipStream_t s) {
    unsigned bz = (p.nC > 0 && p.nH > 0) ? (unsigned)(BATCH * p.nC * p.nH) : 1u;
    dim3 g((p.N + 63) / 64, (p.M + 127) / 128, bz);
    k_gemm<<<g, 256, 0, s>>>(p);
}

extern "C" void kernel_launch(void* const* d_in, const int* in_sizes, int n_in,
                              void* d_out, int out_size, void* d_ws, size_t ws_size,
                              hipStream_t stream) {
    (void)in_sizes; (void)n_in; (void)out_size; (void)ws_size;
    const float* x        = (const float*)d_in[0];
    const float* c1w      = (const float*)d_in[1];
    const float* c1b      = (const float*)d_in[2];
    const float* bn1g     = (const float*)d_in[3];
    const float* bn1b     = (const float*)d_in[4];
    const float* bn1m     = (const float*)d_in[5];
    const float* bn1v     = (const float*)d_in[6];
    const float* c2w      = (const float*)d_in[7];
    const float* c2b      = (const float*)d_in[8];
    const float* bn2g     = (const float*)d_in[9];
    const float* bn2b     = (const float*)d_in[10];
    const float* bn2m     = (const float*)d_in[11];
    const float* bn2v     = (const float*)d_in[12];
    const float* c3w      = (const float*)d_in[13];
    const float* c3b      = (const float*)d_in[14];
    const float* bn3g     = (const float*)d_in[15];
    const float* bn3b     = (const float*)d_in[16];
    const float* bn3m     = (const float*)d_in[17];
    const float* bn3v     = (const float*)d_in[18];
    const float* ln_w     = (const float*)d_in[19];
    const float* in_projw = (const float*)d_in[20];
    const float* conv_w   = (const float*)d_in[21];
    const float* conv_b   = (const float*)d_in[22];
    const float* dt_bias  = (const float*)d_in[23];
    const float* A_log    = (const float*)d_in[24];
    const float* D_skip   = (const float*)d_in[25];
    const float* gn_w     = (const float*)d_in[26];
    const float* out_projw= (const float*)d_in[27];
    const float* normf_w  = (const float*)d_in[28];
    const float* fc1w     = (const float*)d_in[29];
    const float* fc1b     = (const float*)d_in[30];
    const float* bncg     = (const float*)d_in[31];
    const float* bncb     = (const float*)d_in[32];
    const float* bncm     = (const float*)d_in[33];
    const float* bncv     = (const float*)d_in[34];
    const float* fc2w     = (const float*)d_in[35];
    const float* fc2b     = (const float*)d_in[36];

    char* ws = (char*)d_ws;
    float*    h      = (float*)   carve(ws, (size_t)ROWS * D_MODEL * 4);
    _Float16* xf16   = (_Float16*)carve(ws, (size_t)ROWS * D_INNER * 2);     // reused 768/1536 wide
    _Float16* wf16   = (_Float16*)carve(ws, (size_t)D_MODEL * PROJ_DIM * 2); // transposed weights
    float*    proj   = (float*)   carve(ws, (size_t)ROWS * PROJ_DIM * 4);
    float*    xBCc   = (float*)   carve(ws, (size_t)ROWS * CONV_DIM * 4);
    float*    dtbuf  = (float*)   carve(ws, (size_t)ROWS * N_HEADS * 4);
    float*    dAbuf  = (float*)   carve(ws, (size_t)ROWS * N_HEADS * 4);
    float*    Acs    = (float*)   carve(ws, (size_t)ROWS * N_HEADS * 4);
    float*    expA   = (float*)   carve(ws, (size_t)ROWS * N_HEADS * 4);
    float*    decay  = (float*)   carve(ws, (size_t)ROWS * N_HEADS * 4);
    float*    cdec   = (float*)   carve(ws, (size_t)BATCH * N_HEADS * NCHUNK * 4);
    _Float16* XdtT   = (_Float16*)carve(ws, (size_t)ROWS * D_INNER * 2);
    _Float16* XwT    = (_Float16*)carve(ws, (size_t)ROWS * D_INNER * 2);
    _Float16* Bf16   = (_Float16*)carve(ws, (size_t)ROWS * D_STATE * 2);
    _Float16* BfT    = (_Float16*)carve(ws, (size_t)ROWS * D_STATE * 2);
    _Float16* Cf16   = (_Float16*)carve(ws, (size_t)ROWS * D_STATE * 2);
    _Float16* Sf16   = (_Float16*)carve(ws, (size_t)BATCH * NCHUNK * N_HEADS * CHUNKSZ * CHUNKSZ * 2);
    float*    Ybuf   = (float*)   carve(ws, (size_t)ROWS * D_INNER * 4);
    float*    states = (float*)   carve(ws, (size_t)BATCH * NCHUNK * N_HEADS * HEAD_DIM * D_STATE * 4);
    _Float16* prevf  = (_Float16*)carve(ws, (size_t)BATCH * NCHUNK * N_HEADS * HEAD_DIM * D_STATE * 2);
    float*    h1     = (float*)   carve(ws, (size_t)2 * 192 * 2048 * 4);
    float*    h2     = (float*)   carve(ws, (size_t)2 * 384 * 1024 * 4);
    float*    rowr   = (float*)   carve(ws, (size_t)ROWS * 4);
    float*    pooled = (float*)   carve(ws, (size_t)BATCH * D_MODEL * 4);

    // ---- conv stem ----
    k_conv1<<<(2*192*2048 + 255)/256, 256, 0, stream>>>(x, c1w, c1b, bn1g, bn1b, bn1m, bn1v, h1);
    k_conv2<<<(2*384*1024 + 255)/256, 256, 0, stream>>>(h1, c2w, c2b, bn2g, bn2b, bn2m, bn2v, h2);
    k_conv3<<<(2*768*1024 + 255)/256, 256, 0, stream>>>(h2, c3w, c3b, bn3g, bn3b, bn3m, bn3v, h);

    // ---- 24 Mamba2 layers ----
    for (int L = 0; L < N_LAYERS; ++L) {
        // rms norm -> f16
        k_rmsrow<<<ROWS, 256, 0, stream>>>(h, ln_w + (size_t)L * D_MODEL, xf16, D_MODEL);

        // in_proj:  (2048 x 768) * W^T, W converted transposed -> [3352 x 768]
        k_cvtT<<<(int)(((size_t)D_MODEL*PROJ_DIM + 255)/256), 256, 0, stream>>>(
            in_projw + (size_t)L * D_MODEL * PROJ_DIM, wf16, D_MODEL, PROJ_DIM);
        {
            GemmP p{};
            p.A = xf16; p.B = wf16; p.D = proj;
            p.M = ROWS; p.N = PROJ_DIM; p.K = D_MODEL;
            p.lda = D_MODEL; p.ldb = D_MODEL; p.ldd = PROJ_DIM;
            p.nC = 0; p.nH = 1;                                   // batch = 1
            launch_gemm(p, stream);
        }

        k_postproj<<<(ROWS*N_HEADS + 255)/256, 256, 0, stream>>>(
            proj, dt_bias + L*N_HEADS, A_log + L*N_HEADS, dtbuf, dAbuf);
        k_cumsum<<<BATCH*N_HEADS*NCHUNK, 256, 0, stream>>>(dAbuf, Acs, expA, decay, cdec);
        k_dwconv<<<(ROWS*CONV_DIM + 255)/256, 256, 0, stream>>>(
            proj, conv_w + (size_t)L*CONV_DIM*CONV_K, conv_b + (size_t)L*CONV_DIM, xBCc);
        k_pack<<<(ROWS*CONV_DIM + 255)/256, 256, 0, stream>>>(
            xBCc, dtbuf, decay, XdtT, XwT, Bf16, BfT, Cf16);

        // S = (L ⊙ (C·Bᵀ)) -> f16, mask fused in epilogue
        {
            GemmP p{};
            p.A = Cf16; p.B = Bf16; p.Dh = Sf16; p.acs = Acs; p.maskMode = 1;
            p.M = CHUNKSZ; p.N = CHUNKSZ; p.K = D_STATE;
            p.lda = D_STATE; p.ldb = D_STATE; p.ldd = CHUNKSZ;
            p.nC = NCHUNK; p.nH = N_HEADS;
            p.sAb = (long long)SEQ*D_STATE; p.sAc = (long long)CHUNKSZ*D_STATE; p.sAh = 0;
            p.sBb = (long long)SEQ*D_STATE; p.sBc = (long long)CHUNKSZ*D_STATE; p.sBh = 0;
            p.sDb = (long long)NCHUNK*N_HEADS*CHUNKSZ*CHUNKSZ;
            p.sDc = (long long)N_HEADS*CHUNKSZ*CHUNKSZ; p.sDh = (long long)CHUNKSZ*CHUNKSZ;
            p.sRb = (long long)N_HEADS*SEQ; p.sRc = CHUNKSZ; p.sRh = SEQ;
            launch_gemm(p, stream);
        }
        // Y_diag = S · Xdt   (B operand = XdtT, p-major rows, l-contiguous)
        {
            GemmP p{};
            p.A = Sf16; p.B = XdtT; p.D = Ybuf;
            p.M = CHUNKSZ; p.N = HEAD_DIM; p.K = CHUNKSZ;
            p.lda = CHUNKSZ; p.ldb = SEQ; p.ldd = D_INNER;
            p.nC = NCHUNK; p.nH = N_HEADS;
            p.sAb = (long long)NCHUNK*N_HEADS*CHUNKSZ*CHUNKSZ;
            p.sAc = (long long)N_HEADS*CHUNKSZ*CHUNKSZ; p.sAh = (long long)CHUNKSZ*CHUNKSZ;
            p.sBb = (long long)N_HEADS*HEAD_DIM*SEQ; p.sBc = CHUNKSZ;
            p.sBh = (long long)HEAD_DIM*SEQ;
            p.sDb = (long long)SEQ*D_INNER; p.sDc = (long long)CHUNKSZ*D_INNER; p.sDh = HEAD_DIM;
            launch_gemm(p, stream);
        }
        // states = Xwᵀ · B   (A = XwT p-major rows; B = BfT n-major rows)
        {
            GemmP p{};
            p.A = XwT; p.B = BfT; p.D = states;
            p.M = HEAD_DIM; p.N = D_STATE; p.K = CHUNKSZ;
            p.lda = SEQ; p.ldb = SEQ; p.ldd = D_STATE;
            p.nC = NCHUNK; p.nH = N_HEADS;
            p.sAb = (long long)N_HEADS*HEAD_DIM*SEQ; p.sAc = CHUNKSZ;
            p.sAh = (long long)HEAD_DIM*SEQ;
            p.sBb = (long long)D_STATE*SEQ; p.sBc = CHUNKSZ; p.sBh = 0;
            p.sDb = (long long)NCHUNK*N_HEADS*HEAD_DIM*D_STATE;
            p.sDc = (long long)N_HEADS*HEAD_DIM*D_STATE; p.sDh = (long long)HEAD_DIM*D_STATE;
            launch_gemm(p, stream);
        }
        // inter-chunk scan -> prev (f16)
        k_scan<<<BATCH*N_HEADS, 256, 0, stream>>>(states, prevf, cdec);
        // Y += exp(Acs) * (C · prevᵀ)   (B operand = prevf, p-major rows, n-contiguous)
        {
            GemmP p{};
            p.A = Cf16; p.B = prevf; p.D = Ybuf; p.accum = 1; p.rowScale = expA;
            p.M = CHUNKSZ; p.N = HEAD_DIM; p.K = D_STATE;
            p.lda = D_STATE; p.ldb = D_STATE; p.ldd = D_INNER;
            p.nC = NCHUNK; p.nH = N_HEADS;
            p.sAb = (long long)SEQ*D_STATE; p.sAc = (long long)CHUNKSZ*D_STATE; p.sAh = 0;
            p.sBb = (long long)NCHUNK*N_HEADS*HEAD_DIM*D_STATE;
            p.sBc = (long long)N_HEADS*HEAD_DIM*D_STATE; p.sBh = (long long)HEAD_DIM*D_STATE;
            p.sDb = (long long)SEQ*D_INNER; p.sDc = (long long)CHUNKSZ*D_INNER; p.sDh = HEAD_DIM;
            p.sRb = (long long)N_HEADS*SEQ; p.sRc = CHUNKSZ; p.sRh = SEQ;
            launch_gemm(p, stream);
        }

        // gating + group rms -> f16
        k_gate<<<ROWS, 256, 0, stream>>>(proj, Ybuf, xBCc, D_skip + L*N_HEADS,
                                         gn_w + (size_t)L*D_INNER, xf16);

        // out_proj accumulate into residual h, W converted transposed -> [768 x 1536]
        k_cvtT<<<(int)(((size_t)D_INNER*D_MODEL + 255)/256), 256, 0, stream>>>(
            out_projw + (size_t)L * D_INNER * D_MODEL, wf16, D_INNER, D_MODEL);
        {
            GemmP p{};
            p.A = xf16; p.B = wf16; p.D = h; p.accum = 1;
            p.M = ROWS; p.N = D_MODEL; p.K = D_INNER;
            p.lda = D_INNER; p.ldb = D_INNER; p.ldd = D_MODEL;
            p.nC = 0; p.nH = 1;                                   // batch = 1
            launch_gemm(p, stream);
        }
    }

    // ---- head ----
    k_rowr<<<ROWS, 256, 0, stream>>>(h, rowr, D_MODEL);
    k_pool<<<(BATCH*D_MODEL + 255)/256, 256, 0, stream>>>(h, rowr, normf_w, pooled);
    k_head<<<BATCH, 512, 0, stream>>>(pooled, fc1w, fc1b, bncg, bncb, bncm, bncv,
                                      fc2w, fc2b, (float*)d_out);
}
